// CSwinAttention_block_73547019976981
// MI455X (gfx1250) — matrix operand order repaired and checked by reading, not compiled, not run on hardware
//
#include <hip/hip_runtime.h>
#include <hip/hip_bf16.h>

typedef _Float16 half_t;
typedef __attribute__((ext_vector_type(16))) _Float16 v16h;
typedef __attribute__((ext_vector_type(8)))  _Float16 v8h;
typedef __attribute__((ext_vector_type(8)))  float    v8f;

#define SCALE_ 0.08838834764831845f   // (256/2)^-0.5
#define EPS_   1e-4f

// ---- helpers ---------------------------------------------------------------

// B-operand fragment (16x16x32 f16): same K order as ldA
// (lanes 0-15: K=0..7,16..23, lanes 16-31: K=8..15,24..31). p = row base, 16B aligned.
__device__ __forceinline__ v16h ldB(const half_t* p, int hi) {
  union { v16h v; uint4 u[2]; } t;
  t.u[0] = *(const uint4*)(p + hi * 8);
  t.u[1] = *(const uint4*)(p + 16 + hi * 8);
  return t.v;
}

// A-operand fragment (16x16x32 f16): per ISA 7.12.2, within each 16-K group
// lanes 0-15 hold K=g*16+[0..8), lanes 16-31 hold K=g*16+[8..16).
// p = row base at K-chunk start (32 halves wide), 16B aligned.
__device__ __forceinline__ v16h ldA(const half_t* p, int hi) {
  union { v16h v; uint4 u[2]; } t;
  t.u[0] = *(const uint4*)(p + hi * 8);        // K group 0
  t.u[1] = *(const uint4*)(p + 16 + hi * 8);   // K group 1
  return t.v;
}

// async global->LDS copy (16B per lane), tracked by ASYNCcnt
__device__ __forceinline__ void async_ld_b128(unsigned lds_off, const void* gptr) {
  asm volatile("global_load_async_to_lds_b128 %0, %1, off"
               :: "v"(lds_off), "v"(gptr)
               : "memory");
}
__device__ __forceinline__ void wait_async0() {
  asm volatile("s_wait_asynccnt 0" ::: "memory");
}

// window-local token index -> global token index (b*H*W + y*W + x)
__device__ __forceinline__ long tok_of(int branch, int win, int t) {
  if (branch == 0) {               // horizontal strips: h=1,hsp=128,w=32,wsp=4
    int bi = win >> 5, wi = win & 31;
    return (long)bi * 16384 + (long)(t >> 2) * 128 + wi * 4 + (t & 3);
  } else {                         // vertical strips: h=32,hsp=4,w=1,wsp=128
    int bi = win >> 5, st = win & 31;
    return (long)bi * 16384 + (long)st * 512 + t;
  }
}

// ---- weight conversion: f32 -> f16, transpose + channel permutation --------
// wT[n][k] (n in deinterleaved order) = w_qkv[k][orig(n)]
// pT[n][k] = proj_w[n][orig(k)]   (orig = branch*128 + cd*2 + head)
__global__ __launch_bounds__(256) void convert_weights(
    const float* __restrict__ w_qkv, const float* __restrict__ proj_w,
    half_t* __restrict__ wT, half_t* __restrict__ pT) {
  int id = blockIdx.x * 256 + threadIdx.x;          // 262144 total
  if (id < 768 * 256) {
    int n = id >> 8, k = id & 255;
    int part = n >> 8, c = n & 255;
    int br = c >> 7, rr = c & 127;
    int hd = rr >> 6, cdi = rr & 63;
    int oc = part * 256 + br * 128 + cdi * 2 + hd;
    wT[id] = (half_t)w_qkv[(long)k * 768 + oc];
  } else {
    int j = id - 768 * 256;
    int n = j >> 8, k = j & 255;
    int br = k >> 7, rr = k & 127;
    int hd = rr >> 6, cdi = rr & 63;
    int oc = br * 128 + cdi * 2 + hd;
    pT[j] = (half_t)proj_w[(long)n * 256 + oc];
  }
}

// ---- LayerNorm over C=256, x is [B,C,H,W] channel-major --------------------
__global__ __launch_bounds__(256) void ln_kernel(
    const float* __restrict__ x, const float* __restrict__ g,
    const float* __restrict__ bta, half_t* __restrict__ xn) {
  int px = blockIdx.x * 256 + threadIdx.x;          // token id, 65536 total
  int bimg = px >> 14, pix = px & 16383;
  const float* xb = x + (long)bimg * 256 * 16384 + pix;
  float sum = 0.f, sq = 0.f;
  for (int c = 0; c < 256; ++c) {
    float v = xb[(long)c * 16384];
    sum += v; sq += v * v;
  }
  float mean = sum * (1.0f / 256.0f);
  float var  = sq * (1.0f / 256.0f) - mean * mean;
  float rstd = rsqrtf(var + EPS_);
  half_t* orow = xn + (long)px * 256;
  for (int cg = 0; cg < 32; ++cg) {
    v8h v;
#pragma unroll
    for (int j = 0; j < 8; ++j) {
      int c = cg * 8 + j;
      v[j] = (half_t)((xb[(long)c * 16384] - mean) * rstd * g[c] + bta[c]);
    }
    *(v8h*)(orow + cg * 8) = v;
  }
}

// ---- QKV GEMM: [65536,256] x wT^T -> qkvd [65536,768] (deinterleaved) ------
__global__ __launch_bounds__(256) void qkv_gemm(
    const half_t* __restrict__ xn, const half_t* __restrict__ wT,
    half_t* __restrict__ qkvd) {
  const int tid = threadIdx.x;
  const int lane = tid & 31, lm = lane & 15, hi = lane >> 4;
  const int wid = blockIdx.x * 8 + (tid >> 5);      // 24576 waves
  const int nt = wid % 12, mt = wid / 12;
  const int row0 = mt * 32, col0 = nt * 64;

  v8f zero = {};
  v8f acc[2][4];
#pragma unroll
  for (int i = 0; i < 2; ++i)
#pragma unroll
    for (int j = 0; j < 4; ++j) acc[i][j] = zero;

  for (int k0 = 0; k0 < 256; k0 += 32) {
    v16h a[2];
#pragma unroll
    for (int i = 0; i < 2; ++i)
      a[i] = ldA(xn + (long)(row0 + i * 16 + lm) * 256 + k0, hi);
#pragma unroll
    for (int j = 0; j < 4; ++j) {
      v16h b = ldB(wT + (long)(col0 + j * 16 + lm) * 256 + k0, hi);
#pragma unroll
      for (int i = 0; i < 2; ++i)
        acc[i][j] = __builtin_amdgcn_wmma_f32_16x16x32_f16(
            false, a[i], false, b, (short)0, acc[i][j], false, false);
    }
  }
#pragma unroll
  for (int i = 0; i < 2; ++i)
#pragma unroll
    for (int j = 0; j < 4; ++j)
#pragma unroll
      for (int r = 0; r < 8; ++r) {
        int row = row0 + i * 16 + r + hi * 8;
        int col = col0 + j * 16 + lm;
        qkvd[(long)row * 768 + col] = (half_t)acc[i][j][r];
      }
}

// ---- flash attention: block = (branch, window, head, qhalf) ---------------
__global__ __launch_bounds__(256) void attn_kernel(
    const half_t* __restrict__ qkvd, half_t* __restrict__ xc) {
  __shared__ half_t Kb[32][72];    // K block, row-major [key][cd], padded
  __shared__ half_t Vt[64][40];    // V block transposed [cd][key], padded
  __shared__ half_t Pb[8][512];    // per-wave P tile 16x32

  const int bid    = blockIdx.x;
  const int qhalf  = bid & 1;
  const int head   = (bid >> 1) & 1;
  const int win    = (bid >> 2) & 127;
  const int branch = bid >> 9;

  const int tid  = threadIdx.x;
  const int wave = tid >> 5;
  const int lane = tid & 31;
  const int lm   = lane & 15;
  const int hi   = lane >> 4;

  const int qoff = branch * 128 + head * 64;
  const int koff = 256 + qoff;
  const int voff = 512 + qoff;
  const int qr0  = qhalf * 256 + wave * 32;

  // preload Q fragments (A operand, 16 rows x 32 cd per chunk)
  v16h qf[2][2];
#pragma unroll
  for (int mt = 0; mt < 2; ++mt) {
    long t = tok_of(branch, win, qr0 + mt * 16 + lm);
#pragma unroll
    for (int kc = 0; kc < 2; ++kc)
      qf[mt][kc] = ldA(qkvd + t * 768 + qoff + kc * 32, hi);
  }

  v8f zero = {};
  v8f o[2][4];
  float rmax[2][8], rsum[2][8];
#pragma unroll
  for (int mt = 0; mt < 2; ++mt) {
#pragma unroll
    for (int j = 0; j < 4; ++j) o[mt][j] = zero;
#pragma unroll
    for (int r = 0; r < 8; ++r) { rmax[mt][r] = -3.0e38f; rsum[mt][r] = 0.f; }
  }

  for (int kb = 0; kb < 512; kb += 32) {
    __syncthreads();
    { // cooperative K/V block stage: 256 threads x 8 halves each
      int ttok  = tid >> 3;
      int chunk = tid & 7;
      long t = tok_of(branch, win, kb + ttok);
      // K: async global->LDS copy (ASYNCcnt path), row-major
      async_ld_b128((unsigned)(size_t)&Kb[ttok][chunk * 8],
                    qkvd + t * 768 + koff + chunk * 8);
      // V: load + 2-byte transpose scatter into Vt[cd][key]
      union { uint4 u; half_t h[8]; } dv;
      dv.u = *(const uint4*)(qkvd + t * 768 + voff + chunk * 8);
#pragma unroll
      for (int j = 0; j < 8; ++j) Vt[chunk * 8 + j][ttok] = dv.h[j];
      wait_async0();
    }
    __syncthreads();

#pragma unroll
    for (int mt = 0; mt < 2; ++mt) {
      // S = Q * K^T for 16 queries x 32 keys
      v8f sc[2];
#pragma unroll
      for (int nt = 0; nt < 2; ++nt) {
        v8f c = zero;
#pragma unroll
        for (int kc = 0; kc < 2; ++kc) {
          v16h bf = ldB(&Kb[nt * 16 + lm][kc * 32], hi);
          c = __builtin_amdgcn_wmma_f32_16x16x32_f16(
              false, qf[mt][kc], false, bf, (short)0, c, false, false);
        }
        sc[nt] = c;
      }
      // online softmax (rows live in C/D layout: reg r -> row r + hi*8)
#pragma unroll
      for (int r = 0; r < 8; ++r) {
        float s0 = sc[0][r] * SCALE_;
        float s1 = sc[1][r] * SCALE_;
        float mx = fmaxf(s0, s1);
#pragma unroll
        for (int off = 1; off < 16; off <<= 1)
          mx = fmaxf(mx, __shfl_xor(mx, off, 32));
        float mprev = rmax[mt][r];
        float mnew  = fmaxf(mprev, mx);
        rmax[mt][r] = mnew;
        float alpha = __expf(mprev - mnew);
        float p0 = __expf(s0 - mnew);
        float p1 = __expf(s1 - mnew);
        float ps = p0 + p1;
#pragma unroll
        for (int off = 1; off < 16; off <<= 1)
          ps += __shfl_xor(ps, off, 32);
        rsum[mt][r] = rsum[mt][r] * alpha + ps;
#pragma unroll
        for (int j = 0; j < 4; ++j) o[mt][j][r] *= alpha;
        Pb[wave][(r + hi * 8) * 32 + lm]      = (half_t)p0;
        Pb[wave][(r + hi * 8) * 32 + 16 + lm] = (half_t)p1;
      }
      // O += P * V  (A = P 16x32, B = V 32x16 from Vt)
      v16h af = ldA(&Pb[wave][lm * 32], hi);
#pragma unroll
      for (int j = 0; j < 4; ++j) {
        v16h bf = ldB(&Vt[j * 16 + lm][0], hi);
        o[mt][j] = __builtin_amdgcn_wmma_f32_16x16x32_f16(
            false, af, false, bf, (short)0, o[mt][j], false, false);
      }
    }
  }

  // epilogue: O / l  -> xc (deinterleaved channel layout)
#pragma unroll
  for (int mt = 0; mt < 2; ++mt)
#pragma unroll
    for (int r = 0; r < 8; ++r) {
      int qrow = qr0 + mt * 16 + r + hi * 8;
      long t = tok_of(branch, win, qrow);
      float inv = 1.0f / rsum[mt][r];
#pragma unroll
      for (int j = 0; j < 4; ++j) {
        int ch = branch * 128 + head * 64 + j * 16 + lm;
        xc[t * 256 + ch] = (half_t)(o[mt][j][r] * inv);
      }
    }
}

// ---- LePE: depthwise 3x3 (window-local SAME pad) added into xc -------------
__global__ __launch_bounds__(256) void lepe_kernel(
    const half_t* __restrict__ qkvd,
    const float* __restrict__ w1, const float* __restrict__ b1,
    const float* __restrict__ w2, const float* __restrict__ b2,
    half_t* __restrict__ xc) {
  int id = blockIdx.x * 256 + threadIdx.x;  // 2^24 elements
  int xq     = id & 127;
  int yq     = (id >> 7) & 127;
  int cdi    = (id >> 14) & 63;
  int head   = (id >> 20) & 1;
  int branch = (id >> 21) & 1;
  int bimg   = id >> 22;
  const float* wts = branch ? w2 : w1;
  const float* bs  = branch ? b2 : b1;
  int cb = cdi * 2 + head;                  // original interleaved channel
  const float* wp = wts + cb * 9;
  float acc = bs[cb];
  int voff = 512 + branch * 128 + head * 64 + cdi;
#pragma unroll
  for (int dy = -1; dy <= 1; ++dy)
#pragma unroll
    for (int dx = -1; dx <= 1; ++dx) {
      int yy = yq + dy, xx = xq + dx;
      bool ok;
      if (branch == 0)  // x-strip of 4, full height
        ok = (yy >= 0) && (yy < 128) && (xx >= 0) && (xx < 128) &&
             ((xx >> 2) == (xq >> 2));
      else              // y-strip of 4, full width
        ok = (xx >= 0) && (xx < 128) && (yy >= 0) && (yy < 128) &&
             ((yy >> 2) == (yq >> 2));
      if (ok) {
        long t = (long)bimg * 16384 + yy * 128 + xx;
        acc += (float)qkvd[t * 768 + voff] * wp[(dy + 1) * 3 + (dx + 1)];
      }
    }
  int ch = branch * 128 + head * 64 + cdi;
  long idx = ((long)bimg * 16384 + yq * 128 + xq) * 256 + ch;
  xc[idx] = (half_t)((float)xc[idx] + acc);
}

// ---- projection GEMM: out[b,o,h,w] = xc @ pT^T + bias ----------------------
__global__ __launch_bounds__(256) void proj_gemm(
    const half_t* __restrict__ xc, const half_t* __restrict__ pT,
    const float* __restrict__ pb, float* __restrict__ out) {
  const int tid = threadIdx.x;
  const int lane = tid & 31, lm = lane & 15, hi = lane >> 4;
  const int wid = blockIdx.x * 8 + (tid >> 5);      // 8192 waves
  const int nt = wid & 3, mt = wid >> 2;
  const int row0 = mt * 32, col0 = nt * 64;

  v8f zero = {};
  v8f acc[2][4];
#pragma unroll
  for (int i = 0; i < 2; ++i)
#pragma unroll
    for (int j = 0; j < 4; ++j) acc[i][j] = zero;

  for (int k0 = 0; k0 < 256; k0 += 32) {
    v16h a[2];
#pragma unroll
    for (int i = 0; i < 2; ++i)
      a[i] = ldA(xc + (long)(row0 + i * 16 + lm) * 256 + k0, hi);
#pragma unroll
    for (int j = 0; j < 4; ++j) {
      v16h b = ldB(pT + (long)(col0 + j * 16 + lm) * 256 + k0, hi);
#pragma unroll
      for (int i = 0; i < 2; ++i)
        acc[i][j] = __builtin_amdgcn_wmma_f32_16x16x32_f16(
            false, a[i], false, b, (short)0, acc[i][j], false, false);
    }
  }
#pragma unroll
  for (int i = 0; i < 2; ++i)
#pragma unroll
    for (int j = 0; j < 4; ++j)
#pragma unroll
      for (int r = 0; r < 8; ++r) {
        int row = row0 + i * 16 + r + hi * 8;
        int col = col0 + j * 16 + lm;
        int bimg = row >> 14, pix = row & 16383;
        out[((long)bimg * 256 + col) * 16384 + pix] = acc[i][j][r] + pb[col];
      }
}

// ---- launcher --------------------------------------------------------------
extern "C" void kernel_launch(void* const* d_in, const int* in_sizes, int n_in,
                              void* d_out, int out_size, void* d_ws, size_t ws_size,
                              hipStream_t stream) {
  (void)in_sizes; (void)n_in; (void)out_size; (void)ws_size;
  const float* x      = (const float*)d_in[0];
  const float* ln_g   = (const float*)d_in[1];
  const float* ln_b   = (const float*)d_in[2];
  const float* w_qkv  = (const float*)d_in[3];
  const float* lw1    = (const float*)d_in[4];
  const float* lb1    = (const float*)d_in[5];
  const float* lw2    = (const float*)d_in[6];
  const float* lb2    = (const float*)d_in[7];
  const float* proj_w = (const float*)d_in[8];
  const float* proj_b = (const float*)d_in[9];
  float* out = (float*)d_out;

  char* ws = (char*)d_ws;
  half_t* xn   = (half_t*)(ws);                 // 65536*256 f16 = 32 MiB
  half_t* wT   = (half_t*)(ws + 33554432);      // 768*256 f16
  half_t* pT   = (half_t*)(ws + 33947648);      // 256*256 f16
  half_t* qkvd = (half_t*)(ws + 34078720);      // 65536*768 f16 = 96 MiB
  half_t* xc   = (half_t*)(ws + 134742016);     // 65536*256 f16 = 32 MiB

  convert_weights<<<1024, 256, 0, stream>>>(w_qkv, proj_w, wT, pT);
  ln_kernel<<<256, 256, 0, stream>>>(x, ln_g, ln_b, xn);
  qkv_gemm<<<3072, 256, 0, stream>>>(xn, wT, qkvd);
  attn_kernel<<<1024, 256, 0, stream>>>(qkvd, xc);
  lepe_kernel<<<65536, 256, 0, stream>>>(qkvd, lw1, lb1, lw2, lb2, xc);
  proj_gemm<<<1024, 256, 0, stream>>>(xc, pT, proj_b, out);
}
